// MoE_70841190580926
// MI455X (gfx1250) — matrix operand order repaired and checked by reading, not compiled, not run on hardware
//
#include <hip/hip_runtime.h>
#include <hip/hip_bf16.h>

// ---------------------------------------------------------------------------
// MoE top-1:  out[n] = w[n] * (We[e*] @ x[n] + be[e*]),  gate_weights = softmax
// N=8192, D=1024, O=1024, E=16.  Expert GEMM in split-bf16 (hi+lo, 3 WMMAs)
// with f32 accumulation on v_wmma_f32_16x16x32_bf16.
// Per-wave tile 32x32 (4 accumulators) -> 12 WMMAs per 16 b128 loads.
// ---------------------------------------------------------------------------

#define MOE_N 8192
#define MOE_D 1024
#define MOE_O 1024
#define MOE_E 16
#define TILE_M 64
#define TILE_N 128
#define MAX_TILES (MOE_N / TILE_M + MOE_E)   // 144

typedef __attribute__((ext_vector_type(16))) __bf16 v16bf;
typedef __attribute__((ext_vector_type(8)))  float  v8f;

// ---- f32 -> bf16 round-to-nearest-even (bit trick) ----
__device__ __forceinline__ unsigned short f32_to_bf16_rn(float f) {
    unsigned u = __float_as_uint(f);
    unsigned r = u + 0x7FFFu + ((u >> 16) & 1u);
    return (unsigned short)(r >> 16);
}
__device__ __forceinline__ float bf16_bits_to_f32(unsigned short h) {
    return __uint_as_float(((unsigned)h) << 16);
}

// ---------------------------------------------------------------------------
// 0) zero the tiny metadata (counts etc.)
// ---------------------------------------------------------------------------
__global__ void moe_init(int* counts, int* nTiles) {
    if (blockIdx.x == 0 && threadIdx.x < MOE_E) counts[threadIdx.x] = 0;
    if (blockIdx.x == 0 && threadIdx.x == 0) nTiles[0] = 0;
}

// ---------------------------------------------------------------------------
// 1) split f32 -> (bf16 hi, bf16 lo) so that hi+lo ~= f32
// ---------------------------------------------------------------------------
__global__ __launch_bounds__(256) void moe_split(const float* __restrict__ src,
                                                 unsigned short* __restrict__ hi,
                                                 unsigned short* __restrict__ lo,
                                                 int n) {
    int stride = gridDim.x * blockDim.x;
    for (int i = blockIdx.x * blockDim.x + threadIdx.x; i < n; i += stride) {
        float f = src[i];
        unsigned short h = f32_to_bf16_rn(f);
        float residual = f - bf16_bits_to_f32(h);
        hi[i] = h;
        lo[i] = f32_to_bf16_rn(residual);
    }
}

// ---------------------------------------------------------------------------
// 2) gate: logits = x[n]·Wg[e], softmax, top-1 select (exact f32 math)
//    one block per token, 256 threads = 16 experts x 16 partial lanes
// ---------------------------------------------------------------------------
__global__ __launch_bounds__(256) void moe_gate(const float* __restrict__ x,
                                                const float* __restrict__ Wg,
                                                float* __restrict__ gate_out,
                                                int* __restrict__ esel,
                                                float* __restrict__ ewt,
                                                int* __restrict__ counts) {
    __shared__ float xs[MOE_D];
    __shared__ float part[256];
    __shared__ float logits[MOE_E];
    __shared__ float wts[MOE_E];

    int n = blockIdx.x;
    int tid = threadIdx.x;
    const float* xr = x + (size_t)n * MOE_D;
    for (int i = tid; i < MOE_D; i += 256) xs[i] = xr[i];
    __syncthreads();

    int e = tid >> 4, p = tid & 15;
    const float* wr = Wg + (size_t)e * MOE_D;
    float s = 0.f;
    for (int j = p; j < MOE_D; j += 16) s = fmaf(xs[j], wr[j], s);
    part[tid] = s;
    __syncthreads();

    if (tid < MOE_E) {
        float t = 0.f;
        for (int i = 0; i < 16; ++i) t += part[tid * 16 + i];
        logits[tid] = t;
    }
    __syncthreads();

    if (tid < MOE_E) {
        float mx = logits[0];
        for (int i = 1; i < MOE_E; ++i) mx = fmaxf(mx, logits[i]);
        float den = 0.f;
        for (int i = 0; i < MOE_E; ++i) den += __expf(logits[i] - mx);
        float w = __expf(logits[tid] - mx) / den;
        gate_out[(size_t)n * MOE_E + tid] = w;
        wts[tid] = w;
    }
    __syncthreads();

    if (tid == 0) {
        int arg = 0; float best = wts[0];
        for (int i = 1; i < MOE_E; ++i)
            if (wts[i] > best) { best = wts[i]; arg = i; }
        esel[n] = arg;
        ewt[n]  = best;
        atomicAdd(counts + arg, 1);
    }
}

// ---------------------------------------------------------------------------
// 3) scheduler: prefix sums over 16 experts, build 64-token-padded tile table
// ---------------------------------------------------------------------------
__global__ void moe_sched(const int* __restrict__ counts,
                          int* __restrict__ cursor,
                          int* __restrict__ nTiles,
                          int* __restrict__ tileExpert,
                          int* __restrict__ tileStart,
                          int* __restrict__ tileCount) {
    if (blockIdx.x != 0 || threadIdx.x != 0) return;
    int off = 0, nt = 0;
    for (int e = 0; e < MOE_E; ++e) {
        int c = counts[e];
        cursor[e] = off;
        int tiles = (c + TILE_M - 1) / TILE_M;
        for (int t = 0; t < tiles; ++t) {
            tileExpert[nt] = e;
            tileStart[nt]  = off + t * TILE_M;
            int rem = c - t * TILE_M;
            tileCount[nt]  = rem < TILE_M ? rem : TILE_M;
            ++nt;
        }
        off += tiles * TILE_M;
    }
    nTiles[0] = nt;
}

// ---------------------------------------------------------------------------
// 4) scatter tokens into per-expert contiguous lists
// ---------------------------------------------------------------------------
__global__ __launch_bounds__(256) void moe_scatter(const int* __restrict__ esel,
                                                   int* __restrict__ cursor,
                                                   int* __restrict__ tokenList) {
    int n = blockIdx.x * blockDim.x + threadIdx.x;
    if (n >= MOE_N) return;
    int e = esel[n];
    int pos = atomicAdd(cursor + e, 1);
    tokenList[pos] = n;
}

// ---------------------------------------------------------------------------
// WMMA fragment loads (wave32): contiguous k = 16*half + s, same order for A and B.
// A 16x32 bf16: lanes 0-15 -> K 0..15 contiguous; lanes 16-31 -> K 16..31
// B 32x16 bf16: lanes 0-15 -> K 0..15 contiguous; lanes 16-31 -> K 16..31
// ---------------------------------------------------------------------------
__device__ __forceinline__ v16bf load_a_frag(const unsigned short* p, int half) {
    union { v16bf v; uint4 q[2]; } u;
    u.q[0] = *(const uint4*)(p + 16 * half);
    u.q[1] = *(const uint4*)(p + 16 * half + 8);
    return u.v;
}
__device__ __forceinline__ v16bf load_b_frag(const unsigned short* p, int half) {
    union { v16bf v; uint4 q[2]; } u;
    u.q[0] = *(const uint4*)(p + 16 * half);
    u.q[1] = *(const uint4*)(p + 16 * half + 8);
    return u.v;
}
__device__ __forceinline__ v8f wmma_bf16(v16bf a, v16bf b, v8f c) {
    return __builtin_amdgcn_wmma_f32_16x16x32_bf16(
        /*neg_a=*/false, a, /*neg_b=*/false, b,
        /*c_mod=*/(short)0, c, /*reuse_a=*/false, /*reuse_b=*/false);
}

// ---------------------------------------------------------------------------
// 5) expert GEMM: block = 64 tokens x 128 outputs, 8 waves (2m x 4n).
//    Each wave owns a 32x32 slab: 4 accumulator tiles, 12 WMMAs / K=32 step.
//    Split-bf16: acc += Ah*Bh + Ah*Bl + Al*Bh.
// ---------------------------------------------------------------------------
__global__ __launch_bounds__(256) void moe_gemm(
    const unsigned short* __restrict__ xh, const unsigned short* __restrict__ xl,
    const unsigned short* __restrict__ weh, const unsigned short* __restrict__ wel,
    const float* __restrict__ be, const float* __restrict__ ewt,
    const int* __restrict__ tokenList, const int* __restrict__ tileExpert,
    const int* __restrict__ tileStart, const int* __restrict__ tileCount,
    const int* __restrict__ nTiles, float* __restrict__ out) {

    int tile = blockIdx.x;
    if (tile >= nTiles[0]) return;          // uniform: EXEC stays all-1 below

    int e      = tileExpert[tile];
    int tstart = tileStart[tile];
    int tcount = tileCount[tile];
    int nBase  = blockIdx.y * TILE_N;

    int tid  = threadIdx.x;
    int wave = tid >> 5;
    int lane = tid & 31;
    int wm   = wave & 1;                    // 32-row slab of the block tile
    int wn   = wave >> 1;                   // 32-wide column slab
    int l15  = lane & 15;
    int half = lane >> 4;

    // A rows for this lane: m-tiles at 32*wm and 32*wm+16 (clamp padded rows)
    int m0 = 32 * wm + l15;
    int m1 = m0 + 16;
    int tokA0 = tokenList[tstart + (m0 < tcount ? m0 : 0)];
    int tokA1 = tokenList[tstart + (m1 < tcount ? m1 : 0)];
    const unsigned short* a0h = xh + (size_t)tokA0 * MOE_D;
    const unsigned short* a0l = xl + (size_t)tokA0 * MOE_D;
    const unsigned short* a1h = xh + (size_t)tokA1 * MOE_D;
    const unsigned short* a1l = xl + (size_t)tokA1 * MOE_D;

    // B rows (We is [E][O][D], K-major like A -> same chunking)
    int n0 = nBase + 32 * wn + l15;
    const unsigned short* b0h = weh + ((size_t)e * MOE_O + n0) * MOE_D;
    const unsigned short* b0l = wel + ((size_t)e * MOE_O + n0) * MOE_D;
    const unsigned short* b1h = b0h + (size_t)16 * MOE_D;
    const unsigned short* b1l = b0l + (size_t)16 * MOE_D;

    v8f c00 = {}; v8f c01 = {}; v8f c10 = {}; v8f c11 = {};

#pragma unroll 2
    for (int k = 0; k < MOE_D; k += 32) {
        v16bf fa0h = load_a_frag(a0h + k, half);
        v16bf fa0l = load_a_frag(a0l + k, half);
        v16bf fa1h = load_a_frag(a1h + k, half);
        v16bf fa1l = load_a_frag(a1l + k, half);
        v16bf fb0h = load_b_frag(b0h + k, half);
        v16bf fb0l = load_b_frag(b0l + k, half);
        v16bf fb1h = load_b_frag(b1h + k, half);
        v16bf fb1l = load_b_frag(b1l + k, half);

        c00 = wmma_bf16(fa0h, fb0h, c00);
        c01 = wmma_bf16(fa0h, fb1h, c01);
        c10 = wmma_bf16(fa1h, fb0h, c10);
        c11 = wmma_bf16(fa1h, fb1h, c11);

        c00 = wmma_bf16(fa0h, fb0l, c00);
        c01 = wmma_bf16(fa0h, fb1l, c01);
        c10 = wmma_bf16(fa1h, fb0l, c10);
        c11 = wmma_bf16(fa1h, fb1l, c11);

        c00 = wmma_bf16(fa0l, fb0h, c00);
        c01 = wmma_bf16(fa0l, fb1h, c01);
        c10 = wmma_bf16(fa1l, fb0h, c10);
        c11 = wmma_bf16(fa1l, fb1h, c11);
    }

    // Epilogue: out[tok, n] = w_tok * (acc + be[e, n]); guard padded rows.
    union { v8f v; float f[8]; } u00, u01, u10, u11;
    u00.v = c00; u01.v = c01; u10.v = c10; u11.v = c11;
    float bias0 = be[(size_t)e * MOE_O + n0];
    float bias1 = be[(size_t)e * MOE_O + n0 + 16];

#pragma unroll
    for (int r = 0; r < 8; ++r) {
        int m = r + 8 * half;               // C layout: lanes>=16 hold rows 8..15
        // m-tile 0 (rows 32*wm + m)
        {
            int mrow = 32 * wm + m;
            bool valid = mrow < tcount;
            int tok = tokenList[tstart + (valid ? mrow : 0)];
            float w = ewt[tok];
            if (valid) {
                out[(size_t)tok * MOE_O + n0]      = w * (u00.f[r] + bias0);
                out[(size_t)tok * MOE_O + n0 + 16] = w * (u01.f[r] + bias1);
            }
        }
        // m-tile 1 (rows 32*wm + 16 + m)
        {
            int mrow = 32 * wm + 16 + m;
            bool valid = mrow < tcount;
            int tok = tokenList[tstart + (valid ? mrow : 0)];
            float w = ewt[tok];
            if (valid) {
                out[(size_t)tok * MOE_O + n0]      = w * (u10.f[r] + bias0);
                out[(size_t)tok * MOE_O + n0 + 16] = w * (u11.f[r] + bias1);
            }
        }
    }
}

// ---------------------------------------------------------------------------
// launch
// ---------------------------------------------------------------------------
extern "C" void kernel_launch(void* const* d_in, const int* in_sizes, int n_in,
                              void* d_out, int out_size, void* d_ws, size_t ws_size,
                              hipStream_t stream) {
    const float* x  = (const float*)d_in[0];   // [N, D]
    const float* Wg = (const float*)d_in[1];   // [E, D]
    const float* We = (const float*)d_in[2];   // [E, O, D]
    const float* be = (const float*)d_in[3];   // [E, O]

    float* out      = (float*)d_out;                         // [N, O]
    float* gate_out = (float*)d_out + (size_t)MOE_N * MOE_O; // [N, E]

    // workspace carve-up (~96.1 MB)
    char* ws = (char*)d_ws;
    size_t off = 0;
    auto take = [&](size_t bytes) { void* p = ws + off; off = (off + bytes + 255) & ~(size_t)255; return p; };
    unsigned short* xh  = (unsigned short*)take((size_t)MOE_N * MOE_D * 2);
    unsigned short* xl  = (unsigned short*)take((size_t)MOE_N * MOE_D * 2);
    unsigned short* weh = (unsigned short*)take((size_t)MOE_E * MOE_O * MOE_D * 2);
    unsigned short* wel = (unsigned short*)take((size_t)MOE_E * MOE_O * MOE_D * 2);
    int*   esel      = (int*)take(MOE_N * 4);
    float* ewt       = (float*)take(MOE_N * 4);
    int*   counts    = (int*)take(MOE_E * 4);
    int*   cursor    = (int*)take(MOE_E * 4);
    int*   nTiles    = (int*)take(4);
    int*   tileExpert= (int*)take(MAX_TILES * 4);
    int*   tileStart = (int*)take(MAX_TILES * 4);
    int*   tileCount = (int*)take(MAX_TILES * 4);
    int*   tokenList = (int*)take((size_t)MAX_TILES * TILE_M * 4);

    moe_init<<<1, 32, 0, stream>>>(counts, nTiles);

    moe_split<<<4096, 256, 0, stream>>>(x, xh, xl, MOE_N * MOE_D);
    moe_split<<<8192, 256, 0, stream>>>(We, weh, wel, MOE_E * MOE_O * MOE_D);

    moe_gate<<<MOE_N, 256, 0, stream>>>(x, Wg, gate_out, esel, ewt, counts);

    moe_sched<<<1, 1, 0, stream>>>(counts, cursor, nTiles,
                                   tileExpert, tileStart, tileCount);

    moe_scatter<<<MOE_N / 256, 256, 0, stream>>>(esel, cursor, tokenList);

    dim3 grid(MAX_TILES, MOE_O / TILE_N);   // 144 x 8; excess tiles early-exit
    moe_gemm<<<grid, 256, 0, stream>>>(xh, xl, weh, wel, be, ewt,
                                       tokenList, tileExpert, tileStart,
                                       tileCount, nTiles, out);
}
